// AttentionLayer_4879082848963
// MI455X (gfx1250) — matrix operand order repaired and checked by reading, not compiled, not run on hardware
//
#include <hip/hip_runtime.h>
#include <hip/hip_bf16.h>

typedef __attribute__((ext_vector_type(16))) _Float16 v16h;
typedef __attribute__((ext_vector_type(8)))  _Float16 v8h;
typedef __attribute__((ext_vector_type(8)))  float    v8f;
typedef __attribute__((ext_vector_type(4)))  int      v4i;

typedef __attribute__((address_space(1))) v4i gv4i;  // global int4
typedef __attribute__((address_space(3))) v4i lv4i;  // LDS int4

#if defined(__has_builtin)
#if __has_builtin(__builtin_amdgcn_global_load_async_to_lds_b128)
#define USE_ASYNC_LDS 1
#endif
#endif
#ifndef USE_ASYNC_LDS
#define USE_ASYNC_LDS 0
#endif

constexpr int Bb = 2;
constexpr int Ss = 1024;
constexpr int Dd = 1024;
constexpr int Hh = 16;
constexpr int DK = 64;    // Dd / Hh
constexpr int DE = 128;   // Dd / 8
constexpr float SCALE = 0.125f;  // 1/sqrt(64)

constexpr int LDS_STRIDE = 36;   // 36*n mod 64 is a permutation over n=0..15

// ---------------------------------------------------------------------------
// WMMA fragment helpers (contiguous k per lane half, wave32)
//
// A (16x32 f16, MxK): lane = hi*16 + m ; halves[0..15] = A[m, k0+hi*16 + j]
// B (32x16 f16, KxN): lane = khi*16 + n; halves[0..15] = B[k0+khi*16 + j, n]
//                     (for NT gemm, column n of B == row n of W, K contiguous)
// C/D (16x16 f32):    lane = hi*16 + n ; vgpr v = C[v + 8*hi, n]
// ---------------------------------------------------------------------------

__device__ __forceinline__ v8f wmma_f16(v16h a, v16h b, v8f c) {
  return __builtin_amdgcn_wmma_f32_16x16x32_f16(false, a, false, b,
                                                (short)0, c, false, false);
}

__device__ __forceinline__ v16h make_afrag_f32(const float* __restrict__ row,
                                               int k0, int hi) {
  const float* p = row + k0 + hi * 16;
  float4 a0 = *(const float4*)(p);
  float4 a1 = *(const float4*)(p + 4);
  float4 b0 = *(const float4*)(p + 8);
  float4 b1 = *(const float4*)(p + 12);
  v16h f;
  f[0]  = (_Float16)a0.x; f[1]  = (_Float16)a0.y; f[2]  = (_Float16)a0.z; f[3]  = (_Float16)a0.w;
  f[4]  = (_Float16)a1.x; f[5]  = (_Float16)a1.y; f[6]  = (_Float16)a1.z; f[7]  = (_Float16)a1.w;
  f[8]  = (_Float16)b0.x; f[9]  = (_Float16)b0.y; f[10] = (_Float16)b0.z; f[11] = (_Float16)b0.w;
  f[12] = (_Float16)b1.x; f[13] = (_Float16)b1.y; f[14] = (_Float16)b1.z; f[15] = (_Float16)b1.w;
  return f;
}

__device__ __forceinline__ v16h make_afrag_f16(const _Float16* __restrict__ row,
                                               int k0, int hi) {
  const _Float16* p = row + k0 + hi * 16;
  v8h a = *(const v8h*)(p);
  v8h b = *(const v8h*)(p + 8);
  v16h f;
#pragma unroll
  for (int j = 0; j < 8; ++j) { f[j] = a[j]; f[8 + j] = b[j]; }
  return f;
}

__device__ __forceinline__ v16h make_bfrag_f16(const _Float16* __restrict__ col,
                                               int k0, int khi) {
  const _Float16* p = col + k0 + khi * 16;
  v8h a = *(const v8h*)(p);
  v8h b = *(const v8h*)(p + 8);
  v16h f;
#pragma unroll
  for (int j = 0; j < 8; ++j) { f[j] = a[j]; f[8 + j] = b[j]; }
  return f;
}

// B fragment from an LDS-staged f32 tile row (32 k-values, current tile only)
__device__ __forceinline__ v16h make_bfrag_lds(const float* p, int khi) {
  const float* q = p + khi * 16;
  float4 a0 = *(const float4*)(q);
  float4 a1 = *(const float4*)(q + 4);
  float4 b0 = *(const float4*)(q + 8);
  float4 b1 = *(const float4*)(q + 12);
  v16h f;
  f[0]  = (_Float16)a0.x; f[1]  = (_Float16)a0.y; f[2]  = (_Float16)a0.z; f[3]  = (_Float16)a0.w;
  f[4]  = (_Float16)a1.x; f[5]  = (_Float16)a1.y; f[6]  = (_Float16)a1.z; f[7]  = (_Float16)a1.w;
  f[8]  = (_Float16)b0.x; f[9]  = (_Float16)b0.y; f[10] = (_Float16)b0.z; f[11] = (_Float16)b0.w;
  f[12] = (_Float16)b1.x; f[13] = (_Float16)b1.y; f[14] = (_Float16)b1.z; f[15] = (_Float16)b1.w;
  return f;
}

template <int N>
__device__ __forceinline__ void wait_async() {
#if USE_ASYNC_LDS
#if __has_builtin(__builtin_amdgcn_s_wait_asynccnt)
  __builtin_amdgcn_s_wait_asynccnt((short)N);
#else
  if constexpr (N == 0) asm volatile("s_wait_asynccnt 0x0" ::: "memory");
  else                  asm volatile("s_wait_asynccnt 0x4" ::: "memory");
#endif
#endif
}

// ---------------------------------------------------------------------------
// Generic NT GEMM:  out[M,N] = A[M,K] * W[N,K]^T
//   A_HALF:   A is f16 (else f32, converted on load)
//   OUT_MODE: 0 = f32 row-major, 1 = f16 row-major, 2 = f16 transposed [N,M]
//
// Block = 4 waves = 128 threads covering 128(M) x 64(N); each wave owns a
// 32x64 tile (8 accumulators, 2 A-fragments, B shared). The 64x32 f32 W tile
// is staged once per block in LDS, double-buffered via async global->LDS
// (ASYNCcnt) so the next tile streams while WMMAs consume the current one.
// grid = (N/64, M/128).
// ---------------------------------------------------------------------------
template <bool A_HALF, int OUT_MODE>
__global__ __launch_bounds__(128) void gemm_nt(const void* __restrict__ Av,
                                               const float* __restrict__ W,
                                               void* __restrict__ outv,
                                               int M, int N, int K,
                                               int lda, int ldw, int ldo) {
  __shared__ float wt[2][64 * LDS_STRIDE];  // 2 x 9216 B

  const int tid  = threadIdx.x;
  const int lane = tid & 31;
  const int wave = tid >> 5;
  const int mrow = lane & 15;
  const int hi   = lane >> 4;
  const int m0   = blockIdx.y * 128 + wave * 32;
  const int n0   = blockIdx.x * 64;

  const float*    arf0 = (const float*)Av    + (size_t)(m0 + mrow) * lda;
  const float*    arf1 = (const float*)Av    + (size_t)(m0 + 16 + mrow) * lda;
  const _Float16* arh0 = (const _Float16*)Av + (size_t)(m0 + mrow) * lda;
  const _Float16* arh1 = (const _Float16*)Av + (size_t)(m0 + 16 + mrow) * lda;

  // Stage the 64(N) x 32(K) f32 tile of W into LDS: 512 16B segments over
  // 128 threads -> 4 async b128 copies per thread.
  auto load_tile = [&](int buf, int k0) {
#pragma unroll
    for (int s = 0; s < 4; ++s) {
      const int seg = tid + s * 128;
      const int row = seg >> 3;
      const int col = (seg & 7) * 4;
      const float* g = W + (size_t)(n0 + row) * ldw + k0 + col;
      float* l = &wt[buf][row * LDS_STRIDE + col];
#if USE_ASYNC_LDS
      __builtin_amdgcn_global_load_async_to_lds_b128(
          (gv4i*)(void*)g, (lv4i*)(void*)l, 0, 0);
#else
      *(float4*)l = *(const float4*)g;
#endif
    }
  };

  v8f acc0[4] = {};
  v8f acc1[4] = {};

  load_tile(0, 0);
  int buf = 0;
  for (int k0 = 0; k0 < K; k0 += 32) {
    if (k0 + 32 < K) {              // prefetch next tile into other buffer
      load_tile(buf ^ 1, k0 + 32);
      wait_async<4>();              // previous tile's 4 copies complete
    } else {
      wait_async<0>();
    }
    __syncthreads();                // tile visible to all waves

    v16h a0, a1;
    if constexpr (A_HALF) {
      a0 = make_afrag_f16(arh0, k0, hi);
      a1 = make_afrag_f16(arh1, k0, hi);
    } else {
      a0 = make_afrag_f32(arf0, k0, hi);
      a1 = make_afrag_f32(arf1, k0, hi);
    }
    const float* bufp = wt[buf];
#pragma unroll
    for (int nf = 0; nf < 4; ++nf) {
      v16h b = make_bfrag_lds(bufp + (nf * 16 + mrow) * LDS_STRIDE, hi);
      acc0[nf] = wmma_f16(a0, b, acc0[nf]);
      acc1[nf] = wmma_f16(a1, b, acc1[nf]);
    }
    __syncthreads();                // done reading before buffer is reused
    buf ^= 1;
  }

  if constexpr (OUT_MODE == 0) {
    float* out = (float*)outv;
#pragma unroll
    for (int nf = 0; nf < 4; ++nf)
#pragma unroll
      for (int v = 0; v < 8; ++v) {
        const int n = n0 + nf * 16 + mrow;
        out[(size_t)(m0 + 8 * hi + v) * ldo + n]      = acc0[nf][v];
        out[(size_t)(m0 + 16 + 8 * hi + v) * ldo + n] = acc1[nf][v];
      }
  } else if constexpr (OUT_MODE == 1) {
    _Float16* out = (_Float16*)outv;
#pragma unroll
    for (int nf = 0; nf < 4; ++nf)
#pragma unroll
      for (int v = 0; v < 8; ++v) {
        const int n = n0 + nf * 16 + mrow;
        out[(size_t)(m0 + 8 * hi + v) * ldo + n]      = (_Float16)acc0[nf][v];
        out[(size_t)(m0 + 16 + 8 * hi + v) * ldo + n] = (_Float16)acc1[nf][v];
      }
  } else {  // transposed f16: out[n*ldo + m]; M-axis contiguous -> 16B stores
    _Float16* out = (_Float16*)outv;
#pragma unroll
    for (int nf = 0; nf < 4; ++nf) {
      v8h p0, p1;
#pragma unroll
      for (int v = 0; v < 8; ++v) {
        p0[v] = (_Float16)acc0[nf][v];
        p1[v] = (_Float16)acc1[nf][v];
      }
      _Float16* base = out + (size_t)(n0 + nf * 16 + mrow) * ldo + m0 + 8 * hi;
      *(v8h*)(base)      = p0;
      *(v8h*)(base + 16) = p1;
    }
  }
}

// ---------------------------------------------------------------------------
// scores[b,h,q,k] = SCALE * (Q K^T) + 0.1 * (ctx ctx^T)
// One wave computes a 16(q) x 64(k) tile; grid = (S/256, S/16, B*H), 4 waves.
// Written directly into the attn region of d_out (f32), softmaxed in place.
// ---------------------------------------------------------------------------
__global__ __launch_bounds__(128) void scores_kernel(
    const _Float16* __restrict__ Qh, const _Float16* __restrict__ Kh,
    const _Float16* __restrict__ Ctxh, float* __restrict__ attn) {
  const int lane  = threadIdx.x & 31;
  const int wave  = threadIdx.x >> 5;
  const int mrow  = lane & 15;
  const int hi    = lane >> 4;
  const int kbase = blockIdx.x * 256 + wave * 64;
  const int q0    = blockIdx.y * 16;
  const int bh    = blockIdx.z;           // b*H + h
  const int b     = bh >> 4;
  const int h     = bh & 15;

  const _Float16* qrow = Qh   + (size_t)(b * Ss + q0 + mrow) * Dd + h * DK;
  const _Float16* crow = Ctxh + (size_t)(b * Ss + q0 + mrow) * DE;
  const _Float16* krow[4];
  const _Float16* cbrow[4];
#pragma unroll
  for (int nf = 0; nf < 4; ++nf) {
    krow[nf]  = Kh   + (size_t)(b * Ss + kbase + nf * 16 + mrow) * Dd + h * DK;
    cbrow[nf] = Ctxh + (size_t)(b * Ss + kbase + nf * 16 + mrow) * DE;
  }

  v8f aqk[4] = {};
  v8f ainf[4] = {};
#pragma unroll
  for (int k0 = 0; k0 < DK; k0 += 32) {          // Q K^T over head dim (64)
    v16h a = make_afrag_f16(qrow, k0, hi);
#pragma unroll
    for (int nf = 0; nf < 4; ++nf)
      aqk[nf] = wmma_f16(a, make_bfrag_f16(krow[nf], k0, hi), aqk[nf]);
  }
#pragma unroll
  for (int k0 = 0; k0 < DE; k0 += 32) {          // ctx ctx^T over DE (128)
    v16h a = make_afrag_f16(crow, k0, hi);
#pragma unroll
    for (int nf = 0; nf < 4; ++nf)
      ainf[nf] = wmma_f16(a, make_bfrag_f16(cbrow[nf], k0, hi), ainf[nf]);
  }

#pragma unroll
  for (int nf = 0; nf < 4; ++nf)
#pragma unroll
    for (int v = 0; v < 8; ++v)
      attn[((size_t)bh * Ss + q0 + 8 * hi + v) * Ss + kbase + nf * 16 + mrow] =
          aqk[nf][v] * SCALE + 0.1f * ainf[nf][v];
}

// ---------------------------------------------------------------------------
// In-place row softmax over 1024 f32 elements; one block (256 thr) per row.
// ---------------------------------------------------------------------------
__global__ __launch_bounds__(256) void softmax_kernel(float* __restrict__ attn) {
  __shared__ float red[256];
  const int t = threadIdx.x;
  float* p = attn + (size_t)blockIdx.x * Ss;

  float4 v = *(const float4*)(p + t * 4);
  float m = fmaxf(fmaxf(v.x, v.y), fmaxf(v.z, v.w));
  red[t] = m;
  __syncthreads();
#pragma unroll
  for (int s = 128; s > 0; s >>= 1) {
    if (t < s) red[t] = fmaxf(red[t], red[t + s]);
    __syncthreads();
  }
  m = red[0];
  __syncthreads();

  v.x = __expf(v.x - m);
  v.y = __expf(v.y - m);
  v.z = __expf(v.z - m);
  v.w = __expf(v.w - m);
  red[t] = v.x + v.y + v.z + v.w;
  __syncthreads();
#pragma unroll
  for (int s = 128; s > 0; s >>= 1) {
    if (t < s) red[t] += red[t + s];
    __syncthreads();
  }
  const float inv = 1.0f / red[0];
  v.x *= inv; v.y *= inv; v.z *= inv; v.w *= inv;
  *(float4*)(p + t * 4) = v;
}

// ---------------------------------------------------------------------------
// O[b,q,h*64+dk] = attn[b,h,q,:] @ V  (V stored transposed: Vt[d][b*S+s]).
// One wave: 16(q) x 64(dk) tile = the full head dim.  grid = (S/64, B*H).
// ---------------------------------------------------------------------------
__global__ __launch_bounds__(128) void av_kernel(const float* __restrict__ attn,
                                                 const _Float16* __restrict__ Vt,
                                                 _Float16* __restrict__ Oh) {
  const int lane = threadIdx.x & 31;
  const int wave = threadIdx.x >> 5;
  const int mrow = lane & 15;
  const int hi   = lane >> 4;
  const int q0   = (blockIdx.x * 4 + wave) * 16;
  const int bh   = blockIdx.y;
  const int b    = bh >> 4;
  const int h    = bh & 15;

  const float* arow = attn + ((size_t)bh * Ss + q0 + mrow) * Ss;
  const _Float16* vrow[4];
#pragma unroll
  for (int nf = 0; nf < 4; ++nf)
    vrow[nf] = Vt + (size_t)(h * DK + nf * 16 + mrow) * (Bb * Ss) + b * Ss;

  v8f acc[4] = {};
  for (int k0 = 0; k0 < Ss; k0 += 32) {
    v16h a = make_afrag_f32(arow, k0, hi);   // f32 attn -> f16 on the fly
#pragma unroll
    for (int nf = 0; nf < 4; ++nf)
      acc[nf] = wmma_f16(a, make_bfrag_f16(vrow[nf], k0, hi), acc[nf]);
  }

#pragma unroll
  for (int nf = 0; nf < 4; ++nf)
#pragma unroll
    for (int v = 0; v < 8; ++v)
      Oh[(size_t)(b * Ss + q0 + 8 * hi + v) * Dd + h * DK + nf * 16 + mrow] =
          (_Float16)acc[nf][v];
}

// ---------------------------------------------------------------------------
extern "C" void kernel_launch(void* const* d_in, const int* in_sizes, int n_in,
                              void* d_out, int out_size, void* d_ws,
                              size_t ws_size, hipStream_t stream) {
  const float* X  = (const float*)d_in[0];
  const float* Wq = (const float*)d_in[1];
  const float* Wk = (const float*)d_in[2];
  const float* Wv = (const float*)d_in[3];
  const float* Wo = (const float*)d_in[4];
  const float* We = (const float*)d_in[5];

  const int M = Bb * Ss;  // 2048 rows

  // f16 workspace: Qh(4MB) Kh(4MB) Vt(4MB, transposed [D, B*S]) Ctx(0.5MB) Oh(4MB)
  _Float16* Qh   = (_Float16*)d_ws;
  _Float16* Kh   = Qh + (size_t)M * Dd;
  _Float16* Vt   = Kh + (size_t)M * Dd;
  _Float16* Ctxh = Vt + (size_t)M * Dd;
  _Float16* Oh   = Ctxh + (size_t)M * DE;

  float* out  = (float*)d_out;                 // [B,S,D]
  float* attn = out + (size_t)M * Dd;          // [B,H,S,S], scores->softmax in place

  dim3 blk(128);
  // Projections (f32 in, f16 out; V transposed)
  gemm_nt<false, 1><<<dim3(Dd / 64, M / 128), blk, 0, stream>>>(X, Wq, Qh, M, Dd, Dd, Dd, Dd, Dd);
  gemm_nt<false, 1><<<dim3(Dd / 64, M / 128), blk, 0, stream>>>(X, Wk, Kh, M, Dd, Dd, Dd, Dd, Dd);
  gemm_nt<false, 2><<<dim3(Dd / 64, M / 128), blk, 0, stream>>>(X, Wv, Vt, M, Dd, Dd, Dd, Dd, M);
  gemm_nt<false, 1><<<dim3(DE / 64, M / 128), blk, 0, stream>>>(X, We, Ctxh, M, DE, Dd, Dd, Dd, DE);

  // Fused scores = scale*QK^T + 0.1*ctx ctx^T, straight into d_out's attn region
  scores_kernel<<<dim3(Ss / 256, Ss / 16, Bb * Hh), blk, 0, stream>>>(Qh, Kh, Ctxh, attn);

  // Row softmax in place
  softmax_kernel<<<dim3(Bb * Hh * Ss), dim3(256), 0, stream>>>(attn);

  // attn @ V -> Oh, then final projection -> out (f32)
  av_kernel<<<dim3(Ss / 64, Bb * Hh), blk, 0, stream>>>(attn, Vt, Oh);
  gemm_nt<true, 0><<<dim3(Dd / 64, M / 128), blk, 0, stream>>>(Oh, Wo, out, M, Dd, Dd, Dd, Dd, Dd);
}
